// NCA_60928406061712
// MI455X (gfx1250) — matrix operand order repaired and checked by reading, not compiled, not run on hardware
//
#include <hip/hip_runtime.h>
#include <hip/hip_bf16.h>

// ---------------------------------------------------------------------------
// NCA update step, fused for gfx1250 (MI455X):
//   kernel 0: convert W1/W2 to f16 in workspace (B-operand layout = row major [N][K])
//   kernel 1: perceive (depthwise 3x3) + per-pixel MLP via v_wmma_f32_16x16x32_f16
//             + stochastic update; writes x2 to d_out, ch3 plane + prelife to ws
//   kernel 2: life mask from prelife & maxpool(post ch3); zeroes dead pixels in d_out
// ---------------------------------------------------------------------------

#define NB   8
#define NC   32
#define NH   256
#define NW   256
#define HID  256
#define K1   96            // 3*NC
#define PLANE (NH*NW)      // 65536

typedef _Float16 v8h  __attribute__((ext_vector_type(8)));
typedef _Float16 v16h __attribute__((ext_vector_type(16)));
typedef float    v8f  __attribute__((ext_vector_type(8)));
typedef float    v4f  __attribute__((ext_vector_type(4)));

__device__ __forceinline__ v16h frag_pair(const _Float16* lo_p, const _Float16* hi_p) {
    v8h lo = *reinterpret_cast<const v8h*>(lo_p);
    v8h hi = *reinterpret_cast<const v8h*>(hi_p);
    return __builtin_shufflevector(lo, hi, 0,1,2,3,4,5,6,7,8,9,10,11,12,13,14,15);
}

// A fragment (16x32 f16): lane L holds row M=L%16; K groups at +0 and +16 halves.
__device__ __forceinline__ v16h load_afrag(const _Float16* base) {
    return frag_pair(base, base + 16);
}
// B fragment (32x16 f16): lane L holds col N=L%16; K groups at +0 and +16 halves.
__device__ __forceinline__ v16h load_bfrag(const _Float16* base) {
    return frag_pair(base, base + 16);
}

// ---------------------------------------------------------------------------
// Kernel 0: f32 -> f16 weight conversion into workspace
// ---------------------------------------------------------------------------
__global__ __launch_bounds__(256)
void nca_convert_w(const float* __restrict__ W1, const float* __restrict__ W2,
                   _Float16* __restrict__ W1h, _Float16* __restrict__ W2h) {
    int i = blockIdx.x * 256 + threadIdx.x;
    if (i < HID * K1) {
        W1h[i] = (_Float16)W1[i];            // [N=256][K=96], already N-major
    } else if (i < HID * K1 + NC * HID) {
        int j = i - HID * K1;
        W2h[j] = (_Float16)W2[j];            // [N=32][K=256]
    }
}

// ---------------------------------------------------------------------------
// Kernel 1: fused perceive + MLP (WMMA) + stochastic update.
// 128 threads = 4 waves; each wave owns one 16-pixel row strip.
// ---------------------------------------------------------------------------
__global__ __launch_bounds__(128)
void nca_main(const float* __restrict__ xg, const float* __restrict__ randg,
              const float* __restrict__ b1g,
              const _Float16* __restrict__ W1h, const _Float16* __restrict__ W2h,
              float* __restrict__ outg, float* __restrict__ ch3g,
              unsigned char* __restrict__ preg) {
    // per-wave LDS tiles (padded rows so 16 M-rows hit distinct bank groups)
    __shared__ alignas(16) _Float16 pbuf[4][16][104];   // [M=16][K=96(+pad)]
    __shared__ alignas(16) _Float16 hbuf[4][16][264];   // [M=16][K=256(+pad)]
    __shared__ float firebuf[4][16];

    const int tid  = threadIdx.x;
    const int wave = tid >> 5;
    const int lane = tid & 31;
    const int mrow = lane & 15;     // M row within strip
    const int g    = lane >> 4;     // lane-half (K-group select)

    const int strip = blockIdx.x * 4 + wave;            // 32768 strips total
    const int x0 = (strip & 15) << 4;
    const int y  = (strip >> 4) & (NH - 1);
    const int b  = strip >> 12;

    // ---------------- phase 1: perceive -> pbuf, fire, prelife ----------------
    {
        const int gx = x0 + mrow;
        const bool xm = gx > 0, xp = gx < NW - 1;
        const bool ym = y > 0,  yp = y < NH - 1;
        const int  ic = y * NW + gx;
        _Float16* prow = &pbuf[wave][mrow][0];
        #pragma unroll 4
        for (int i = 0; i < 16; ++i) {
            const int c = g + 2 * i;                     // lane-halves cover even/odd c
            const float* xc = xg + (size_t)(b * NC + c) * PLANE;
            const float vcc = xc[ic];
            const float vtl = (ym && xm) ? xc[ic - NW - 1] : 0.f;
            const float vtc =  ym        ? xc[ic - NW]     : 0.f;
            const float vtr = (ym && xp) ? xc[ic - NW + 1] : 0.f;
            const float vcl =  xm        ? xc[ic - 1]      : 0.f;
            const float vcr =  xp        ? xc[ic + 1]      : 0.f;
            const float vbl = (yp && xm) ? xc[ic + NW - 1] : 0.f;
            const float vbc =  yp        ? xc[ic + NW]     : 0.f;
            const float vbr = (yp && xp) ? xc[ic + NW + 1] : 0.f;
            const float sx = (vtr - vtl + 2.f * (vcr - vcl) + vbr - vbl) * 0.125f;
            const float sy = (vbl - vtl + 2.f * (vbc - vtc) + vbr - vtr) * 0.125f;
            prow[3 * c + 0] = (_Float16)vcc;
            prow[3 * c + 1] = (_Float16)sx;
            prow[3 * c + 2] = (_Float16)sy;
        }
        if (lane < 16) {            // one lane per pixel: fire mask + pre-life
            const float ru = randg[(b * NH + y) * NW + gx];
            firebuf[wave][lane] = (ru <= 0.5f) ? 1.f : 0.f;
            const float* x3 = xg + (size_t)(b * NC + 3) * PLANE;
            float mx = x3[ic];
            if (ym)       mx = fmaxf(mx, x3[ic - NW]);
            if (yp)       mx = fmaxf(mx, x3[ic + NW]);
            if (xm)       mx = fmaxf(mx, x3[ic - 1]);
            if (xp)       mx = fmaxf(mx, x3[ic + 1]);
            if (ym && xm) mx = fmaxf(mx, x3[ic - NW - 1]);
            if (ym && xp) mx = fmaxf(mx, x3[ic - NW + 1]);
            if (yp && xm) mx = fmaxf(mx, x3[ic + NW - 1]);
            if (yp && xp) mx = fmaxf(mx, x3[ic + NW + 1]);
            preg[(b * NH + y) * NW + gx] = (mx > 0.1f) ? 1 : 0;
        }
    }
    // per-wave LDS producer/consumer: DS ops are in-order within a wave.

    // ---------------- phase 2: GEMM1  h = relu(p @ W1^T + b1) ----------------
    const v16h a0 = load_afrag(&pbuf[wave][mrow][0  + g * 8]);
    const v16h a1 = load_afrag(&pbuf[wave][mrow][32 + g * 8]);
    const v16h a2 = load_afrag(&pbuf[wave][mrow][64 + g * 8]);

    #pragma unroll 4
    for (int nt = 0; nt < 16; ++nt) {
        const float bv = b1g[nt * 16 + mrow];            // C/D layout: N = lane%16
        v8f acc = {bv, bv, bv, bv, bv, bv, bv, bv};
        const _Float16* w = W1h + (size_t)(nt * 16 + mrow) * K1 + g * 8;
        acc = __builtin_amdgcn_wmma_f32_16x16x32_f16(false, a0, false, load_bfrag(w),      (short)0, acc, false, false);
        acc = __builtin_amdgcn_wmma_f32_16x16x32_f16(false, a1, false, load_bfrag(w + 32), (short)0, acc, false, false);
        acc = __builtin_amdgcn_wmma_f32_16x16x32_f16(false, a2, false, load_bfrag(w + 64), (short)0, acc, false, false);
        #pragma unroll
        for (int r = 0; r < 8; ++r) {                    // D: M = r + 8*g
            const float hv = fmaxf(acc[r], 0.f);
            hbuf[wave][g * 8 + r][nt * 16 + mrow] = (_Float16)hv;
        }
    }

    // ---------------- phase 3: GEMM2  upd = h @ W2^T ----------------
    v16h ha[8];
    #pragma unroll
    for (int kc = 0; kc < 8; ++kc)
        ha[kc] = load_afrag(&hbuf[wave][mrow][kc * 32 + g * 8]);

    v8f u0 = {0.f, 0.f, 0.f, 0.f, 0.f, 0.f, 0.f, 0.f};
    v8f u1 = u0;
    const _Float16* w2a = W2h + (size_t)(mrow)      * HID + g * 8;
    const _Float16* w2b = W2h + (size_t)(16 + mrow) * HID + g * 8;
    #pragma unroll
    for (int kc = 0; kc < 8; ++kc)
        u0 = __builtin_amdgcn_wmma_f32_16x16x32_f16(false, ha[kc], false, load_bfrag(w2a + kc * 32), (short)0, u0, false, false);
    #pragma unroll
    for (int kc = 0; kc < 8; ++kc)
        u1 = __builtin_amdgcn_wmma_f32_16x16x32_f16(false, ha[kc], false, load_bfrag(w2b + kc * 32), (short)0, u1, false, false);

    // ---------------- phase 4: x2 = x + upd*fire -> d_out (+ ch3 aux) --------
    #pragma unroll
    for (int nt2 = 0; nt2 < 2; ++nt2) {
        const v8f u = nt2 ? u1 : u0;
        const int ch = nt2 * 16 + mrow;                  // channel = N
        const size_t base = ((size_t)(b * NC + ch) * NH + y) * NW + x0 + g * 8;
        const v4f xlo = *reinterpret_cast<const v4f*>(xg + base);
        const v4f xhi = *reinterpret_cast<const v4f*>(xg + base + 4);
        v4f olo, ohi;
        #pragma unroll
        for (int r = 0; r < 4; ++r) {
            olo[r] = xlo[r] + u[r]     * firebuf[wave][g * 8 + r];
            ohi[r] = xhi[r] + u[4 + r] * firebuf[wave][g * 8 + 4 + r];
        }
        *reinterpret_cast<v4f*>(outg + base)     = olo;
        *reinterpret_cast<v4f*>(outg + base + 4) = ohi;
        if (ch == 3) {                                   // post-update ch3 plane for pass 2
            const int pb = (b * NH + y) * NW + x0 + g * 8;
            #pragma unroll
            for (int r = 0; r < 4; ++r) { ch3g[pb + r] = olo[r]; ch3g[pb + 4 + r] = ohi[r]; }
        }
    }
}

// ---------------------------------------------------------------------------
// Kernel 2: life = pre & (maxpool3x3(post ch3) > 0.1); zero dead pixels in-place
// ---------------------------------------------------------------------------
__global__ __launch_bounds__(256)
void nca_life(const float* __restrict__ ch3g, const unsigned char* __restrict__ preg,
              float* __restrict__ outg) {
    const int idx = blockIdx.x * 256 + threadIdx.x;
    if (idx >= NB * NH * NW) return;
    const int xx = idx & (NW - 1);
    const int y  = (idx >> 8) & (NH - 1);
    const int b  = idx >> 16;
    float mx = -3.402823466e38f;                         // -inf padding (matches maxpool)
    #pragma unroll
    for (int dy = -1; dy <= 1; ++dy) {
        const int iy = y + dy;
        if (iy < 0 || iy >= NH) continue;
        #pragma unroll
        for (int dx = -1; dx <= 1; ++dx) {
            const int ix = xx + dx;
            if (ix < 0 || ix >= NW) continue;
            mx = fmaxf(mx, ch3g[(b * NH + iy) * NW + ix]);
        }
    }
    const bool alive = (preg[idx] != 0) && (mx > 0.1f);
    if (!alive) {
        float* p = outg + (size_t)b * NC * PLANE + y * NW + xx;
        #pragma unroll
        for (int c = 0; c < NC; ++c) p[(size_t)c * PLANE] = 0.f;
    }
}

// ---------------------------------------------------------------------------
extern "C" void kernel_launch(void* const* d_in, const int* in_sizes, int n_in,
                              void* d_out, int out_size, void* d_ws, size_t ws_size,
                              hipStream_t stream) {
    (void)in_sizes; (void)n_in; (void)out_size; (void)ws_size;
    const float* x    = (const float*)d_in[0];
    const float* ru   = (const float*)d_in[1];
    const float* W1   = (const float*)d_in[2];
    const float* b1   = (const float*)d_in[3];
    const float* W2   = (const float*)d_in[4];
    float* out        = (float*)d_out;

    // workspace layout
    char* ws = (char*)d_ws;
    _Float16* W1h      = (_Float16*)(ws);                      // 48 KB
    _Float16* W2h      = (_Float16*)(ws + 49152);              // 16 KB
    float*    ch3      = (float*)   (ws + 65536);              // 2 MB
    unsigned char* pre = (unsigned char*)(ws + 65536 + (size_t)NB * PLANE * 4); // 512 KB

    // 0: weight conversion (32768 elements)
    nca_convert_w<<<128, 256, 0, stream>>>(W1, W2, W1h, W2h);
    // 1: fused perceive + WMMA MLP + update (32768 strips / 4 waves per block)
    nca_main<<<8192, 128, 0, stream>>>(x, ru, b1, W1h, W2h, out, ch3, pre);
    // 2: alive masking
    nca_life<<<(NB * NH * NW + 255) / 256, 256, 0, stream>>>(ch3, pre, out);
}
